// MultiHeadLatentAttention_42193758716043
// MI455X (gfx1250) — compile-verified, matrix operand order checked by reading, not run on hardware
//
#include <hip/hip_runtime.h>
#include <hip/hip_bf16.h>

typedef _Float16 f16;
typedef __attribute__((ext_vector_type(16))) _Float16 v16h;
typedef __attribute__((ext_vector_type(8)))  _Float16 v8h;
typedef __attribute__((ext_vector_type(8)))  float    v8f;

#define B_  2
#define S_  2048
#define E_  2048
#define H_  16
#define HD_ 128
#define L_  512

// ---------------- wave-level WMMA helpers (wave32, CDNA5) ----------------

__device__ __forceinline__ v8f wmma16(v16h a, v16h b, v8f c) {
  // D = A(16x32 f16) * B(32x16 f16) + C(16x16 f32)
  return __builtin_amdgcn_wmma_f32_16x16x32_f16(false, a, false, b, (short)0, c,
                                                false, false);
}

// Load a 16x32 f16 A-tile (or B-tile from B^T stored row-major) per the ISA
// VGPR layout: lane<16 -> row=lane, K=[0..7]+[16..23]; lane>=16 -> K=[8..15]+[24..31].
__device__ __forceinline__ v16h load_tile16(const f16* base, int ld) {
  int lane = threadIdx.x & 31;
  int row  = lane & 15;
  int ko   = (lane >> 4) << 3;           // 0 or 8
  const f16* p = base + (size_t)row * ld + ko;
  v8h lo = *(const v8h*)(p);             // K = ko .. ko+7      (VGPRs 0..3)
  v8h hi = *(const v8h*)(p + 16);        // K = ko+16 .. ko+23  (VGPRs 4..7)
  return __builtin_shufflevector(lo, hi, 0,1,2,3,4,5,6,7,8,9,10,11,12,13,14,15);
}

// C-layout: VGPR r -> row (r + 8*(lane>=16)), col = lane&15
__device__ __forceinline__ void store_c_f16(v8f acc, f16* base, int ld) {
  int lane = threadIdx.x & 31;
  int col  = lane & 15;
  int rb   = (lane >> 4) << 3;
#pragma unroll
  for (int r = 0; r < 8; ++r)
    base[(size_t)(rb + r) * ld + col] = (f16)acc[r];
}

// CDNA5 async global->LDS copy (ASYNCcnt); 16 bytes per issue.
__device__ __forceinline__ void async_cp16(uint32_t lds_off, const f16* gsrc) {
  asm volatile("global_load_async_to_lds_b128 %0, %1, off"
               :: "v"(lds_off), "v"(gsrc) : "memory");
}
__device__ __forceinline__ void wait_async0() {
  asm volatile("s_wait_asynccnt 0x0" ::: "memory");
}

// ---------------- conversion kernels ----------------

__global__ void k_convert_x(const float* __restrict__ x, f16* __restrict__ xh) {
  int i = blockIdx.x * blockDim.x + threadIdx.x;        // over B*S*E
  if (i >= B_ * S_ * E_) return;
  int e = i % E_;
  int s = (i / E_) % S_;
  int b = i / (E_ * S_);
  int h = e / HD_, d = e % HD_;
  xh[(((size_t)(b * H_ + h)) * S_ + s) * HD_ + d] = (f16)x[i];
}

__global__ void k_f32_to_f16(const float* __restrict__ in, f16* __restrict__ out, int n) {
  int i = blockIdx.x * blockDim.x + threadIdx.x;
  if (i < n) out[i] = (f16)in[i];
}

__global__ void k_transpose_wuv(const float* __restrict__ wuv, f16* __restrict__ wuvT) {
  int i = blockIdx.x * blockDim.x + threadIdx.x;        // over H*L*HD
  if (i >= H_ * L_ * HD_) return;
  int d = i % HD_;
  int l = (i / HD_) % L_;
  int h = i / (HD_ * L_);
  wuvT[((size_t)h * HD_ + d) * L_ + l] = (f16)wuv[i];
}

// ---------------- absorbedT[h][l][i] = sum_j Wuk[h][l][j] * Wq[h][i][j] -----

__global__ void __launch_bounds__(128)
k_absorbed(const f16* __restrict__ wuk, const f16* __restrict__ wq,
           f16* __restrict__ absT) {
  int wid = (blockIdx.x * blockDim.x + threadIdx.x) >> 5;   // 512 waves
  int lt = wid & 31;                 // 32 tiles of 16 rows over L
  int h  = wid >> 5;
  const f16* A  = wuk + ((size_t)h * L_ + lt * 16) * HD_;   // [16 x 128], ld=HD
  const f16* Bt = wq  + (size_t)h * HD_ * HD_;              // B^T rows (n=i), ld=HD
  v8f acc[8] = {};
  for (int k = 0; k < HD_; k += 32) {
    v16h a = load_tile16(A + k, HD_);
#pragma unroll
    for (int n = 0; n < 8; ++n) {
      v16h b = load_tile16(Bt + (size_t)(n * 16) * HD_ + k, HD_);
      acc[n] = wmma16(a, b, acc[n]);
    }
  }
  f16* out = absT + ((size_t)h * L_ + lt * 16) * HD_;
#pragma unroll
  for (int n = 0; n < 8; ++n) store_c_f16(acc[n], out + n * 16, HD_);
}

// ---------------- c = LayerNorm(xh @ Wdkv^T); write f32 to d_out, f16 to ws ---

#define LPAD 520

__global__ void __launch_bounds__(128)
k_c_ln(const f16* __restrict__ xh, const f16* __restrict__ wdkv,
       const float* __restrict__ ln_g, const float* __restrict__ ln_b,
       f16* __restrict__ c16, float* __restrict__ c_out) {
  __shared__ float cs[16][LPAD];
  int blk = blockIdx.x;                       // B*H*(S/16)
  int st = blk % (S_ / 16);
  int h  = (blk / (S_ / 16)) % H_;
  int b  = blk / ((S_ / 16) * H_);
  int wave = threadIdx.x >> 5;                // 4 waves: 128 cols each

  const f16* A  = xh + (((size_t)(b * H_ + h)) * S_ + st * 16) * HD_;
  const f16* Bt = wdkv + (size_t)h * L_ * HD_ + (size_t)(wave * 128) * HD_;
  v8f acc[8] = {};
  for (int k = 0; k < HD_; k += 32) {
    v16h a = load_tile16(A + k, HD_);
#pragma unroll
    for (int n = 0; n < 8; ++n) {
      v16h bt = load_tile16(Bt + (size_t)(n * 16) * HD_ + k, HD_);
      acc[n] = wmma16(a, bt, acc[n]);
    }
  }
  int lane = threadIdx.x & 31;
  int col  = lane & 15;
  int rb   = (lane >> 4) << 3;
#pragma unroll
  for (int n = 0; n < 8; ++n)
#pragma unroll
    for (int r = 0; r < 8; ++r)
      cs[rb + r][wave * 128 + n * 16 + col] = acc[n][r];
  __syncthreads();

  // LayerNorm: 8 threads per row, 64 cols each (xor-reduce within 8 lanes)
  int row = threadIdx.x >> 3;
  int seg = threadIdx.x & 7;
  float sum = 0.f, sq = 0.f;
  for (int j = 0; j < 64; ++j) {
    float v = cs[row][seg * 64 + j];
    sum += v; sq += v * v;
  }
  for (int m = 1; m < 8; m <<= 1) {
    sum += __shfl_xor(sum, m, 32);
    sq  += __shfl_xor(sq,  m, 32);
  }
  float mu  = sum * (1.f / L_);
  float var = sq * (1.f / L_) - mu * mu;
  float rs  = rsqrtf(var + 1e-5f);
  size_t cbase = (((size_t)(b * H_ + h)) * S_ + st * 16 + row) * L_;
  for (int j = 0; j < 64; ++j) {
    int l = seg * 64 + j;
    float v = (cs[row][l] - mu) * rs * ln_g[h * L_ + l] + ln_b[h * L_ + l];
    c_out[cbase + l] = v;
    c16[cbase + l]  = (f16)v;
  }
}

// ---------------- q_lat = (xh @ absorbed) * 1/sqrt(L), f16 ----------------
// 32-row M tile per wave: two A tiles share every B tile (2x L2 reuse).

__global__ void __launch_bounds__(128)
k_qlat(const f16* __restrict__ xh, const f16* __restrict__ absT,
       f16* __restrict__ q16) {
  int wid = (blockIdx.x * blockDim.x + threadIdx.x) >> 5;   // 8192 waves
  int nt = wid & 3;                       // 128-col chunk of L
  int st = (wid >> 2) & 63;               // 32-row chunk of S
  int h  = (wid >> 8) & 15;
  int b  = wid >> 12;
  const f16* A  = xh + (((size_t)(b * H_ + h)) * S_ + st * 32) * HD_;
  const f16* Bt = absT + (size_t)h * L_ * HD_ + (size_t)(nt * 128) * HD_; // rows l
  v8f acc0[8] = {}, acc1[8] = {};
  for (int k = 0; k < HD_; k += 32) {
    v16h a0 = load_tile16(A + k, HD_);
    v16h a1 = load_tile16(A + (size_t)16 * HD_ + k, HD_);
#pragma unroll
    for (int n = 0; n < 8; ++n) {
      v16h bt = load_tile16(Bt + (size_t)(n * 16) * HD_ + k, HD_);
      acc0[n] = wmma16(a0, bt, acc0[n]);
      acc1[n] = wmma16(a1, bt, acc1[n]);
    }
  }
  const float sc = 0.04419417382415922f;  // 1/sqrt(512)
  f16* out = q16 + (((size_t)(b * H_ + h)) * S_ + st * 32) * L_ + nt * 128;
  int lane = threadIdx.x & 31;
  int col = lane & 15, rb = (lane >> 4) << 3;
#pragma unroll
  for (int n = 0; n < 8; ++n)
#pragma unroll
    for (int r = 0; r < 8; ++r) {
      out[(size_t)(rb + r) * L_ + n * 16 + col]        = (f16)(acc0[n][r] * sc);
      out[(size_t)(16 + rb + r) * L_ + n * 16 + col]   = (f16)(acc1[n][r] * sc);
    }
}

// ---------------- vT[b][h][d][t] = sum_l WuvT[h][d][l] * c[b][h][t][l] ------

__global__ void __launch_bounds__(128)
k_vt(const f16* __restrict__ wuvT, const f16* __restrict__ c16,
     f16* __restrict__ vT) {
  int wid = (blockIdx.x * blockDim.x + threadIdx.x) >> 5;   // 2048 waves
  int tt = wid & 15;                 // 128-wide chunk over t (S)
  int dt = (wid >> 4) & 3;           // 32-row chunk over d (HD)
  int h  = (wid >> 6) & 15;
  int b  = wid >> 10;
  const f16* A  = wuvT + ((size_t)h * HD_ + dt * 32) * L_;             // ld=L
  const f16* Bt = c16 + (((size_t)(b * H_ + h)) * S_ + tt * 128) * L_; // rows t
  v8f acc0[8] = {}, acc1[8] = {};
  for (int k = 0; k < L_; k += 32) {
    v16h a0 = load_tile16(A + k, L_);
    v16h a1 = load_tile16(A + (size_t)16 * L_ + k, L_);
#pragma unroll
    for (int n = 0; n < 8; ++n) {
      v16h bt = load_tile16(Bt + (size_t)(n * 16) * L_ + k, L_);
      acc0[n] = wmma16(a0, bt, acc0[n]);
      acc1[n] = wmma16(a1, bt, acc1[n]);
    }
  }
  f16* out = vT + (((size_t)(b * H_ + h)) * HD_ + dt * 32) * S_ + tt * 128;
#pragma unroll
  for (int n = 0; n < 8; ++n) {
    store_c_f16(acc0[n], out + n * 16, S_);
    store_c_f16(acc1[n], out + (size_t)16 * S_ + n * 16, S_);
  }
}

// ---------------- flash attention -----------------------------------------
// 64-query super-tile per block (4 waves x 16 rows, same b,h). The block
// cooperatively double-buffers each 32-key K block (32KB) and V panel (8KB)
// into LDS via global_load_async_to_lds_b128 (ASYNCcnt), so the 4 waves share
// one L2 fetch instead of four. Q tiles are register-resident.

#define KLD 520     // K row stride in LDS (halves): 512 + 8 pad (bank spread)
#define VLD 40      // V row stride in LDS (halves): 32 + 8 pad

__device__ __forceinline__ void stage_k(f16* dst, const f16* src, int tid) {
  uint32_t l0 = (uint32_t)(uintptr_t)dst;
#pragma unroll
  for (int i = 0; i < 16; ++i) {          // 32 rows x 512 halves = 2048 x 16B
    int c   = i * 128 + tid;
    int row = c >> 6;                     // 64 chunks per row
    int cc  = c & 63;
    async_cp16(l0 + (uint32_t)row * (KLD * 2) + (uint32_t)cc * 16u,
               src + (size_t)row * L_ + cc * 8);
  }
}

__device__ __forceinline__ void stage_v(f16* dst, const f16* src, int t0, int tid) {
  uint32_t l0 = (uint32_t)(uintptr_t)dst + (uint32_t)tid * (VLD * 2);
  const f16* g = src + (size_t)tid * S_ + t0;   // row d=tid, 32 cols = 64B
#pragma unroll
  for (int i = 0; i < 4; ++i)
    async_cp16(l0 + (uint32_t)i * 16u, g + i * 8);
}

__global__ void __launch_bounds__(128)
k_attn(const f16* __restrict__ q16, const f16* __restrict__ c16,
       const f16* __restrict__ vT, f16* __restrict__ oh16) {
  __shared__ __align__(16) f16 kbuf[2][32 * KLD];
  __shared__ __align__(16) f16 vbuf[2][128 * VLD];
  __shared__ __align__(16) f16 pst[4][16 * 32];
  int tid  = threadIdx.x;
  int wave = tid >> 5;
  int blk  = blockIdx.x;                 // 1024 blocks: B*H*(S/64)
  int qsuper = blk & 31;
  int h  = (blk >> 5) & 15;
  int b  = blk >> 9;
  int lane = tid & 31;
  int col = lane & 15;
  int rb  = (lane >> 4) << 3;
  int q0 = qsuper * 64 + wave * 16;

  const f16* Q  = q16 + (((size_t)(b * H_ + h)) * S_ + q0) * L_;
  const f16* Kg = c16 + ((size_t)(b * H_ + h)) * S_ * L_;
  const f16* Vg = vT  + ((size_t)(b * H_ + h)) * HD_ * S_;   // [d][t], ld=S
  f16* Po = pst[wave];

  // Q tile resident in registers: 16 k-steps of the scores GEMM
  v16h qa[16];
#pragma unroll
  for (int k = 0; k < 16; ++k) qa[k] = load_tile16(Q + k * 32, L_);

  float m[8], l[8];
  v8f o[8] = {};
#pragma unroll
  for (int r = 0; r < 8; ++r) { m[r] = -1e30f; l[r] = 0.f; }

  int nblk_own = (q0 + 15) / 32 + 1;
  int nblk_max = qsuper * 2 + 2;         // causal range of the last wave

  stage_k(kbuf[0], Kg, tid);             // preload block 0
  stage_v(vbuf[0], Vg, 0, tid);
  wait_async0();
  __syncthreads();

  for (int j = 0; j < nblk_max; ++j) {
    int t0 = j * 32;
    int cur = j & 1, nxt = cur ^ 1;
    if (j + 1 < nblk_max) {              // stream next block while computing
      stage_k(kbuf[nxt], Kg + (size_t)(t0 + 32) * L_, tid);
      stage_v(vbuf[nxt], Vg, t0 + 32, tid);
    }
    if (j < nblk_own) {                  // wave-uniform guard (EXEC stays full)
      const f16* Kb = kbuf[cur];
      v8f s0 = {}, s1 = {};
#pragma unroll
      for (int k = 0; k < 16; ++k) {     // scores, K-dim = 512, from LDS
        v16h b0 = load_tile16(Kb + k * 32, KLD);
        v16h b1 = load_tile16(Kb + (size_t)16 * KLD + k * 32, KLD);
        s0 = wmma16(qa[k], b0, s0);
        s1 = wmma16(qa[k], b1, s1);
      }
      if (t0 + 31 > q0) {                // causal mask (partial block)
#pragma unroll
        for (int r = 0; r < 8; ++r) {
          int row = q0 + rb + r;
          s0[r] = (t0 + col      > row) ? -1e30f : s0[r];
          s1[r] = (t0 + 16 + col > row) ? -1e30f : s1[r];
        }
      }
#pragma unroll
      for (int r = 0; r < 8; ++r) {      // online softmax per row
        float mx = fmaxf(s0[r], s1[r]);
        mx = fmaxf(mx, __shfl_xor(mx, 1, 32));
        mx = fmaxf(mx, __shfl_xor(mx, 2, 32));
        mx = fmaxf(mx, __shfl_xor(mx, 4, 32));
        mx = fmaxf(mx, __shfl_xor(mx, 8, 32));
        float nm = fmaxf(m[r], mx);
        float cf = __expf(m[r] - nm);
        m[r] = nm;
        float p0 = __expf(s0[r] - nm);
        float p1 = __expf(s1[r] - nm);
        float rsum = p0 + p1;
        rsum += __shfl_xor(rsum, 1, 32);
        rsum += __shfl_xor(rsum, 2, 32);
        rsum += __shfl_xor(rsum, 4, 32);
        rsum += __shfl_xor(rsum, 8, 32);
        l[r] = l[r] * cf + rsum;
#pragma unroll
        for (int n = 0; n < 8; ++n) o[n][r] *= cf;
        Po[(rb + r) * 32 + col]      = (f16)p0;   // stage P -> A layout
        Po[(rb + r) * 32 + 16 + col] = (f16)p1;
      }
      v16h pa = load_tile16(Po, 32);     // P as A (16x32)
#pragma unroll
      for (int n = 0; n < 8; ++n) {      // O += P @ V (V panel from LDS)
        v16h bt = load_tile16(vbuf[cur] + (size_t)(n * 16) * VLD, VLD);
        o[n] = wmma16(pa, bt, o[n]);
      }
    }
    wait_async0();                       // next buffers resident
    __syncthreads();                     // everyone done with cur before reuse
  }

  float inv[8];
#pragma unroll
  for (int r = 0; r < 8; ++r) inv[r] = 1.f / l[r];
  f16* out = oh16 + (((size_t)(b * H_ + h)) * S_ + q0) * HD_;
#pragma unroll
  for (int n = 0; n < 8; ++n)
#pragma unroll
    for (int r = 0; r < 8; ++r)
      out[(size_t)(rb + r) * HD_ + n * 16 + col] = (f16)(o[n][r] * inv[r]);
}

// ---------------- per-head output projection: oh @ Wo_h^T -> o16[B,S,E] ----

__global__ void __launch_bounds__(128)
k_oh2(const f16* __restrict__ oh16, const f16* __restrict__ woh16,
      f16* __restrict__ o16) {
  int wid = (blockIdx.x * blockDim.x + threadIdx.x) >> 5;  // 2048 waves
  int qt = wid & 63;                  // 32-row chunk over S
  int h  = (wid >> 6) & 15;
  int b  = wid >> 10;
  const f16* A  = oh16 + (((size_t)(b * H_ + h)) * S_ + qt * 32) * HD_;
  const f16* Bt = woh16 + (size_t)h * HD_ * HD_;           // rows e (Wo_h[h][e][d])
  v8f acc0[8] = {}, acc1[8] = {};
  for (int k = 0; k < HD_; k += 32) {
    v16h a0 = load_tile16(A + k, HD_);
    v16h a1 = load_tile16(A + (size_t)16 * HD_ + k, HD_);
#pragma unroll
    for (int n = 0; n < 8; ++n) {
      v16h bt = load_tile16(Bt + (size_t)(n * 16) * HD_ + k, HD_);
      acc0[n] = wmma16(a0, bt, acc0[n]);
      acc1[n] = wmma16(a1, bt, acc1[n]);
    }
  }
  f16* out = o16 + (((size_t)b * S_ + qt * 32)) * E_ + h * HD_;
#pragma unroll
  for (int n = 0; n < 8; ++n) {
    store_c_f16(acc0[n], out + n * 16, E_);
    store_c_f16(acc1[n], out + (size_t)16 * E_ + n * 16, E_);
  }
}

// ---------------- final: out = o @ Wo^T + bo (f32 to d_out) ----------------

__global__ void __launch_bounds__(128)
k_final(const f16* __restrict__ o16, const f16* __restrict__ wo16,
        const float* __restrict__ bo, float* __restrict__ out) {
  int wid = (blockIdx.x * blockDim.x + threadIdx.x) >> 5;  // 2048 waves
  int nt = wid & 15;                   // 128-col chunk over E
  int mt = wid >> 4;                   // 32-row chunk over B*S (0..127)
  const f16* A  = o16 + (size_t)mt * 32 * E_;
  const f16* Bt = wo16 + (size_t)(nt * 128) * E_;          // rows e (Wo[e][e'])
  v8f acc0[8] = {}, acc1[8] = {};
  for (int k = 0; k < E_; k += 32) {
    v16h a0 = load_tile16(A + k, E_);
    v16h a1 = load_tile16(A + (size_t)16 * E_ + k, E_);
#pragma unroll
    for (int n = 0; n < 8; ++n) {
      v16h bt = load_tile16(Bt + (size_t)(n * 16) * E_ + k, E_);
      acc0[n] = wmma16(a0, bt, acc0[n]);
      acc1[n] = wmma16(a1, bt, acc1[n]);
    }
  }
  float* op = out + (size_t)mt * 32 * E_ + nt * 128;
  int lane = threadIdx.x & 31;
  int col = lane & 15, rb = (lane >> 4) << 3;
#pragma unroll
  for (int n = 0; n < 8; ++n)
#pragma unroll
    for (int r = 0; r < 8; ++r) {
      float bias = bo[nt * 128 + n * 16 + col];
      op[(size_t)(rb + r) * E_ + n * 16 + col]        = acc0[n][r] + bias;
      op[(size_t)(16 + rb + r) * E_ + n * 16 + col]   = acc1[n][r] + bias;
    }
}

// ---------------- host launcher ----------------

extern "C" void kernel_launch(void* const* d_in, const int* in_sizes, int n_in,
                              void* d_out, int out_size, void* d_ws, size_t ws_size,
                              hipStream_t stream) {
  const float* x    = (const float*)d_in[0];
  const float* Wq   = (const float*)d_in[1];
  const float* Wdkv = (const float*)d_in[2];
  const float* Wuk  = (const float*)d_in[3];
  const float* Wuv  = (const float*)d_in[4];
  const float* ln_g = (const float*)d_in[5];
  const float* ln_b = (const float*)d_in[6];
  const float* Wo_h = (const float*)d_in[7];
  const float* Wo   = (const float*)d_in[8];
  const float* bo   = (const float*)d_in[9];

  float* out0 = (float*)d_out;                       // [B,S,E]
  float* cout = out0 + (size_t)B_ * S_ * E_;         // [B,H,S,L]

  char* w = (char*)d_ws;
  size_t off = 0;
  auto take = [&](size_t nhalf) -> f16* {
    f16* p = (f16*)(w + off);
    off = (off + nhalf * 2 + 255) & ~(size_t)255;
    return p;
  };
  f16* xh16   = take((size_t)B_ * H_ * S_ * HD_);
  f16* wq16   = take((size_t)H_ * HD_ * HD_);
  f16* wdkv16 = take((size_t)H_ * L_ * HD_);
  f16* wuk16  = take((size_t)H_ * L_ * HD_);
  f16* wuvT16 = take((size_t)H_ * HD_ * L_);
  f16* woh16  = take((size_t)H_ * HD_ * HD_);
  f16* wo16   = take((size_t)E_ * E_);
  f16* absT16 = take((size_t)H_ * L_ * HD_);
  f16* c16    = take((size_t)B_ * H_ * S_ * L_);
  f16* q16    = take((size_t)B_ * H_ * S_ * L_);
  f16* vT16   = take((size_t)B_ * H_ * HD_ * S_);
  f16* oh16   = take((size_t)B_ * H_ * S_ * HD_);
  f16* o16    = take((size_t)B_ * S_ * E_);

  // conversions
  k_convert_x<<<(B_ * S_ * E_) / 256, 256, 0, stream>>>(x, xh16);
  k_f32_to_f16<<<(H_ * HD_ * HD_ + 255) / 256, 256, 0, stream>>>(Wq, wq16, H_ * HD_ * HD_);
  k_f32_to_f16<<<(H_ * L_ * HD_ + 255) / 256, 256, 0, stream>>>(Wdkv, wdkv16, H_ * L_ * HD_);
  k_f32_to_f16<<<(H_ * L_ * HD_ + 255) / 256, 256, 0, stream>>>(Wuk, wuk16, H_ * L_ * HD_);
  k_f32_to_f16<<<(H_ * HD_ * HD_ + 255) / 256, 256, 0, stream>>>(Wo_h, woh16, H_ * HD_ * HD_);
  k_f32_to_f16<<<(E_ * E_ + 255) / 256, 256, 0, stream>>>(Wo, wo16, E_ * E_);
  k_transpose_wuv<<<(H_ * L_ * HD_ + 255) / 256, 256, 0, stream>>>(Wuv, wuvT16);

  // pipeline
  k_absorbed<<<128, 128, 0, stream>>>(wuk16, wq16, absT16);
  k_c_ln<<<B_ * H_ * (S_ / 16), 128, 0, stream>>>(xh16, wdkv16, ln_g, ln_b, c16, cout);
  k_qlat<<<2048, 128, 0, stream>>>(xh16, absT16, q16);
  k_vt<<<512, 128, 0, stream>>>(wuvT16, c16, vT16);
  k_attn<<<1024, 128, 0, stream>>>(q16, c16, vT16, oh16);
  k_oh2<<<512, 128, 0, stream>>>(oh16, woh16, o16);
  k_final<<<512, 128, 0, stream>>>(o16, wo16, bo, out0);
}
